// LongformerBlock_29403346109028
// MI455X (gfx1250) — matrix operand order repaired and checked by reading, not compiled, not run on hardware
//
#include <hip/hip_runtime.h>

typedef __attribute__((ext_vector_type(16))) _Float16 v16h;
typedef __attribute__((ext_vector_type(8)))  float    v8f;
typedef __attribute__((ext_vector_type(4)))  unsigned u32x4;
typedef __attribute__((ext_vector_type(8)))  int      i32x8;
typedef __attribute__((ext_vector_type(4)))  int      i32x4;

constexpr int kB  = 2;
constexpr int kS  = 2048;
constexpr int kE  = 768;
constexpr int kH  = 12;
constexpr int kDH = 64;
constexpr int kW  = 256;
constexpr int kG  = 16;
constexpr int kFF = 3072;
constexpr int kC  = 8;           // kS / kW
constexpr float NEGF = -1.0e30f;

union V16H { v16h v; _Float16 h[16]; };

__device__ __forceinline__ v8f wmma_f16(v16h a, v16h b, v8f c) {
  return __builtin_amdgcn_wmma_f32_16x16x32_f16(false, a, false, b, (short)0, c, false, false);
}

// Low 32 bits of a generic pointer to LDS == wave-relative LDS byte offset
// (shared aperture base has zero low 32 bits, ISA 10.2).
__device__ __forceinline__ unsigned lds_off(const void* p) {
  return (unsigned)(unsigned long long)p;
}

// Tensor Data Mover: DMA a 2D tile (tile_w elems x tile_h rows, f16, row stride
// ld elems) from global memory into LDS. Descriptor per CDNA5 ISA ch.8.
__device__ __forceinline__ void tdm_load_2d_f16(unsigned dst_lds, const void* gsrc,
                                                int tile_w, int tile_h, int ld) {
  unsigned long long ga = (unsigned long long)gsrc;
  u32x4 g0;
  g0.x = 1u;                                              // count=1, user descriptor
  g0.y = dst_lds;                                         // lds_addr
  g0.z = (unsigned)ga;                                    // global_addr[31:0]
  g0.w = (unsigned)((ga >> 32) & 0x1FFFFFFull) | 0x80000000u; // addr[56:32] | type=2
  i32x8 g1;
  g1[0] = 1 << 16;                                        // data_size = 2 bytes
  g1[1] = (tile_w & 0xFFFF) << 16;                        // tensor_dim0[15:0]
  g1[2] = ((unsigned)tile_w >> 16) | ((tile_h & 0xFFFF) << 16); // td0 hi | td1 lo
  g1[3] = ((unsigned)tile_h >> 16) | ((tile_w & 0xFFFF) << 16); // td1 hi | tile_dim0
  g1[4] = tile_h & 0xFFFF;                                // tile_dim1 (tile_dim2=0)
  g1[5] = ld;                                             // tensor_dim0_stride[31:0]
  g1[6] = 0;
  g1[7] = 0;
  i32x4 z4 = {0, 0, 0, 0};
#if defined(__clang_major__) && (__clang_major__ >= 23)
  i32x8 z8 = {0, 0, 0, 0, 0, 0, 0, 0};
  __builtin_amdgcn_tensor_load_to_lds(g0, g1, z4, z4, z8, 0);
#else
  __builtin_amdgcn_tensor_load_to_lds(g0, g1, z4, z4, 0);
#endif
}

// A fragment (16x32, f16) from row-major src (16 rows, leading dim ld).
__device__ __forceinline__ v16h load_A16x32(const _Float16* src, int ld) {
  int lane = threadIdx.x & 31;
  int m  = lane & 15;
  int kb = (lane & 16) ? 16 : 0;
  V16H r;
#pragma unroll
  for (int i = 0; i < 8; ++i) {
    int k = kb + i * 2;
    r.h[2 * i]     = src[m * ld + k];
    r.h[2 * i + 1] = src[m * ld + k + 1];
  }
  return r.v;
}

// B fragment (32x16): B[k][n] = src[k*ld + n]  (src row-major K x N)
__device__ __forceinline__ v16h load_B_rm(const _Float16* src, int ld) {
  int lane = threadIdx.x & 31;
  int n  = lane & 15;
  int kb = (lane & 16) ? 16 : 0;
  V16H r;
#pragma unroll
  for (int i = 0; i < 8; ++i) {
    int k = kb + i * 2;
    r.h[2 * i]     = src[k * ld + n];
    r.h[2 * i + 1] = src[(k + 1) * ld + n];
  }
  return r.v;
}

// B fragment (32x16) where B[k][n] = src[n*ld + k]  (src row-major N x K)
__device__ __forceinline__ v16h load_B_tr(const _Float16* src, int ld) {
  int lane = threadIdx.x & 31;
  int n  = lane & 15;
  int kb = (lane & 16) ? 16 : 0;
  V16H r;
#pragma unroll
  for (int i = 0; i < 8; ++i) {
    int k = kb + i * 2;
    r.h[2 * i]     = src[n * ld + k];
    r.h[2 * i + 1] = src[n * ld + k + 1];
  }
  return r.v;
}

// ---------------------------------------------------------------- prep ------
__global__ __launch_bounds__(256)
void prep_kernel(const int* __restrict__ am, const int* __restrict__ gm,
                 float* __restrict__ t, int* __restrict__ removed, int* __restrict__ masked) {
  __shared__ int sums[256];
  int b = blockIdx.x, tid = threadIdx.x;
  int local[8];
  int acc = 0;
#pragma unroll
  for (int j = 0; j < 8; ++j) {
    int s = tid * 8 + j;
    int merged = am[b * kS + s] * (gm[b * kS + s] + 1);
    removed[b * kS + s] = (merged != 1) ? 1 : 0;
    masked[b * kS + s]  = (merged == 0) ? 1 : 0;
    acc += (merged == 2) ? 1 : 0;
    local[j] = acc;
  }
  sums[tid] = acc;
  __syncthreads();
  if (tid == 0) {
    int run = 0;
    for (int i = 0; i < 256; ++i) { int v = sums[i]; sums[i] = run; run += v; }
  }
  __syncthreads();
  int base = sums[tid];
#pragma unroll
  for (int j = 0; j < 8; ++j) {
    int s = tid * 8 + j;
    t[b * kS + s] = masked[b * kS + s] ? 0.f : (float)(base + local[j]);
  }
}

// ---------------------------------------------------------------- convert ---
__global__ __launch_bounds__(256)
void cvt_kernel(const float* __restrict__ src, _Float16* __restrict__ dst, int n) {
  int i = blockIdx.x * 256 + threadIdx.x;
  if (i < n) dst[i] = (_Float16)src[i];
}

// ---------------------------------------------------------------- GEMM ------
// C[M,N] = A[M,K] @ B[K,N]; EPI==0: f32 out (+bias); EPI==2: silu(A@B0)*(A@B1) -> f16 out
// A/B tiles staged into LDS by the Tensor Data Mover (wave 0 issues, TENSORcnt).
template <int EPI>
__global__ __launch_bounds__(256)
void gemm_wmma(const _Float16* __restrict__ A, const _Float16* __restrict__ B0,
               const _Float16* __restrict__ B1, const float* __restrict__ bias,
               float* __restrict__ Cf, _Float16* __restrict__ Ch,
               int M, int N, int K) {
  __shared__ _Float16 As[128][32];
  __shared__ _Float16 Bs0[32][128];
  __shared__ _Float16 Bs1[32][128];
  const int bm = blockIdx.x, bn = blockIdx.y;
  const int tid = threadIdx.x, wave = tid >> 5, lane = tid & 31;
  const int wm = wave & 3, wn = wave >> 2;        // 4x2 wave grid, 32x64 per wave
  const int row0 = bm * 128, col0 = bn * 128;
  const v8f z8 = {0.f, 0.f, 0.f, 0.f, 0.f, 0.f, 0.f, 0.f};
  v8f acc0[2][4], acc1[2][4];
#pragma unroll
  for (int i = 0; i < 2; ++i)
#pragma unroll
    for (int j = 0; j < 4; ++j) { acc0[i][j] = z8; acc1[i][j] = z8; }

  const unsigned ldsA  = lds_off(&As[0][0]);
  const unsigned ldsB0 = lds_off(&Bs0[0][0]);
  const unsigned ldsB1 = lds_off(&Bs1[0][0]);

  for (int k0 = 0; k0 < K; k0 += 32) {
    __syncthreads();                              // all waves done with previous tiles
    if (wave == 0) {
      tdm_load_2d_f16(ldsA,  A  + (size_t)row0 * K + k0,   32, 128, K);
      tdm_load_2d_f16(ldsB0, B0 + (size_t)k0 * N + col0,  128,  32, N);
      if (EPI == 2)
        tdm_load_2d_f16(ldsB1, B1 + (size_t)k0 * N + col0, 128, 32, N);
      __builtin_amdgcn_s_wait_tensorcnt(0);
    }
    if (k0 + 32 < K) {                            // warm L2 for next tiles
      __builtin_prefetch(A + (size_t)(row0 + (tid & 127)) * K + k0 + 32, 0, 1);
      __builtin_prefetch(B0 + (size_t)(k0 + 32 + (tid & 31)) * N + col0, 0, 1);
    }
    __syncthreads();                              // tiles visible to all waves

    v16h af0 = load_A16x32(&As[wm * 32][0], 32);
    v16h af1 = load_A16x32(&As[wm * 32 + 16][0], 32);
#pragma unroll
    for (int nt = 0; nt < 4; ++nt) {
      v16h bf = load_B_rm(&Bs0[0][wn * 64 + nt * 16], 128);
      acc0[0][nt] = wmma_f16(af0, bf, acc0[0][nt]);
      acc0[1][nt] = wmma_f16(af1, bf, acc0[1][nt]);
      if (EPI == 2) {
        v16h bg = load_B_rm(&Bs1[0][wn * 64 + nt * 16], 128);
        acc1[0][nt] = wmma_f16(af0, bg, acc1[0][nt]);
        acc1[1][nt] = wmma_f16(af1, bg, acc1[1][nt]);
      }
    }
  }

  const int hf = lane >> 4, nl = lane & 15;
#pragma unroll
  for (int mt = 0; mt < 2; ++mt)
#pragma unroll
    for (int nt = 0; nt < 4; ++nt)
#pragma unroll
      for (int r = 0; r < 8; ++r) {
        int row = row0 + wm * 32 + mt * 16 + r + hf * 8;
        int col = col0 + wn * 64 + nt * 16 + nl;
        float v = acc0[mt][nt][r];
        if (EPI == 0) {
          if (bias) v += bias[col];
          Cf[(size_t)row * N + col] = v;
        } else {
          float g = v / (1.f + __expf(-v));       // silu
          Ch[(size_t)row * N + col] = (_Float16)(g * acc1[mt][nt][r]);
        }
      }
}

// ---------------------------------------------------------------- pack ------
__global__ __launch_bounds__(256)
void pack_kernel(const float* __restrict__ src, const float* __restrict__ t,
                 _Float16* __restrict__ dst, int do_rope, float scale) {
  int i = blockIdx.x * 256 + threadIdx.x;
  if (i >= kB * kS * kE) return;
  int e = i % kE;
  int bs = i / kE;
  int s = bs % kS, b = bs / kS;
  int h = e / kDH, d = e % kDH;
  float val;
  if (do_rope) {
    const int hd = kDH / 2;
    int dd = d % hd;
    float inv = __expf(-(float)dd * (9.210340371976184f / (float)hd)); // 10000^{-dd/hd}
    float ang = t[bs] * inv;
    float cs = __cosf(ang), sn = __sinf(ang);
    float x1 = src[(size_t)bs * kE + h * kDH + dd];
    float x2 = src[(size_t)bs * kE + h * kDH + dd + hd];
    val = (d < hd) ? (x1 * cs - x2 * sn) : (x2 * cs + x1 * sn);
  } else {
    val = src[i];
  }
  val *= scale;
  dst[(((size_t)b * kH + h) * kS + s) * kDH + d] = (_Float16)val;
}

// ---------------------------------------------------------------- band attn -
__global__ __launch_bounds__(256)
void band_attn_kernel(const _Float16* __restrict__ qf, const _Float16* __restrict__ kf,
                      const _Float16* __restrict__ vf, const int* __restrict__ removed,
                      const int* __restrict__ masked, float* __restrict__ attn) {
  __shared__ _Float16 Qs[kW * kDH];      // 32 KB
  __shared__ _Float16 Ks[32 * kDH];      // 4 KB
  __shared__ _Float16 Vs[32 * kDH];      // 4 KB
  __shared__ _Float16 Ps[8][16 * 32];    // 8 KB, per-wave P staging
  __shared__ float rm[32];

  const int c = blockIdx.x, h = blockIdx.y, b = blockIdx.z;
  const int tid = threadIdx.x, wave = tid >> 5, lane = tid & 31;
  const int nl = lane & 15, hf = lane >> 4;
  const size_t headBase = ((size_t)b * kH + h) * kS * kDH;
  const v8f z8 = {0.f, 0.f, 0.f, 0.f, 0.f, 0.f, 0.f, 0.f};

  {
    const _Float16* src = qf + headBase + (size_t)c * kW * kDH;  // contiguous 256x64
#pragma unroll
    for (int i = 0; i < 8; ++i) {
      int cc = tid + i * 256;
      *(uint4*)&Qs[cc * 8] = *(const uint4*)(src + cc * 8);
    }
  }
  __syncthreads();

  v16h qa[2][2];
#pragma unroll
  for (int rt = 0; rt < 2; ++rt) {
    qa[rt][0] = load_A16x32(&Qs[(wave * 32 + rt * 16) * kDH], kDH);
    qa[rt][1] = load_A16x32(&Qs[(wave * 32 + rt * 16) * kDH + 32], kDH);
  }

  float mrow[2][8], lrow[2][8];
  v8f o[2][4];
#pragma unroll
  for (int rt = 0; rt < 2; ++rt) {
#pragma unroll
    for (int r = 0; r < 8; ++r) { mrow[rt][r] = NEGF; lrow[rt][r] = 0.f; }
#pragma unroll
    for (int nt = 0; nt < 4; ++nt) o[rt][nt] = z8;
  }

  for (int it = 0; it < 25; ++it) {
    __syncthreads();
    {
      int row = tid >> 3, pp = tid & 7;  // 32 rows x 8 chunks of 8 halfs
      long grow;
      bool valid;
      if (it == 0) { grow = row; valid = (row < kG); }
      else { grow = (long)(c - 1) * kW + (it - 1) * 32 + row; valid = (grow >= 0) && (grow < kS); }
      uint4 zz = {0u, 0u, 0u, 0u};
      if (valid) {
        *(uint4*)&Ks[row * kDH + pp * 8] = *(const uint4*)(kf + headBase + (size_t)grow * kDH + pp * 8);
        *(uint4*)&Vs[row * kDH + pp * 8] = *(const uint4*)(vf + headBase + (size_t)grow * kDH + pp * 8);
      } else {
        *(uint4*)&Ks[row * kDH + pp * 8] = zz;
        *(uint4*)&Vs[row * kDH + pp * 8] = zz;
      }
      if (tid < 32) {
        float rmv;
        if (it == 0) rmv = (tid < kG) ? 0.f : 1.f;
        else {
          long g = (long)(c - 1) * kW + (it - 1) * 32 + tid;
          rmv = (g < 0 || g >= kS || removed[(size_t)b * kS + g]) ? 1.f : 0.f;
        }
        rm[tid] = rmv;
      }
    }
    __syncthreads();

    float rm0 = rm[nl], rm1 = rm[nl + 16];

#pragma unroll
    for (int rt = 0; rt < 2; ++rt) {
      v8f c0 = z8, c1 = z8;
      c0 = wmma_f16(qa[rt][0], load_B_tr(&Ks[0], kDH), c0);
      c0 = wmma_f16(qa[rt][1], load_B_tr(&Ks[32], kDH), c0);
      c1 = wmma_f16(qa[rt][0], load_B_tr(&Ks[16 * kDH], kDH), c1);
      c1 = wmma_f16(qa[rt][1], load_B_tr(&Ks[16 * kDH + 32], kDH), c1);

#pragma unroll
      for (int r = 0; r < 8; ++r) {
        int prow = wave * 32 + rt * 16 + r + hf * 8;   // query row within chunk
        float s0 = c0[r], s1 = c1[r];
        if (it == 0) {
          s1 = NEGF;                                   // only 16 global columns
        } else {
          int col0 = (it - 1) * 32 + nl;
          int col1 = col0 + 16;
          if (col0 < prow || col0 > prow + 2 * kW || rm0 != 0.f) s0 = NEGF;
          if (col1 < prow || col1 > prow + 2 * kW || rm1 != 0.f) s1 = NEGF;
        }
        float mx = fmaxf(s0, s1);
#pragma unroll
        for (int off = 1; off < 16; off <<= 1) mx = fmaxf(mx, __shfl_xor(mx, off, 32));
        float mnew  = fmaxf(mrow[rt][r], mx);
        float alpha = __expf(mrow[rt][r] - mnew);
        float p0 = __expf(s0 - mnew);
        float p1 = __expf(s1 - mnew);
        float ps = p0 + p1;
#pragma unroll
        for (int off = 1; off < 16; off <<= 1) ps += __shfl_xor(ps, off, 32);
        lrow[rt][r] = lrow[rt][r] * alpha + ps;
        mrow[rt][r] = mnew;
        Ps[wave][(r + hf * 8) * 32 + nl]      = (_Float16)p0;
        Ps[wave][(r + hf * 8) * 32 + nl + 16] = (_Float16)p1;
#pragma unroll
        for (int nt = 0; nt < 4; ++nt) o[rt][nt][r] *= alpha;
      }
      asm volatile("s_wait_dscnt 0" ::: "memory");
      v16h pf = load_A16x32(&Ps[wave][0], 32);
#pragma unroll
      for (int nt = 0; nt < 4; ++nt)
        o[rt][nt] = wmma_f16(pf, load_B_rm(&Vs[nt * 16], kDH), o[rt][nt]);
    }
  }

#pragma unroll
  for (int rt = 0; rt < 2; ++rt)
#pragma unroll
    for (int nt = 0; nt < 4; ++nt)
#pragma unroll
      for (int r = 0; r < 8; ++r) {
        int row = wave * 32 + rt * 16 + r + hf * 8;
        int s = c * kW + row;
        float denom = lrow[rt][r];
        float val = (denom > 0.f) ? o[rt][nt][r] / denom : 0.f;
        if (masked[(size_t)b * kS + s]) val = 0.f;
        attn[((size_t)b * kS + s) * kE + h * kDH + nt * 16 + nl] = val;
      }
}

// ---------------------------------------------------------------- global ----
__global__ __launch_bounds__(32)
void global_attn_kernel(const _Float16* __restrict__ gq, const _Float16* __restrict__ gk,
                        const _Float16* __restrict__ gv, const int* __restrict__ masked,
                        float* __restrict__ attn) {
  __shared__ _Float16 Ps[16 * 32];
  const int bh = blockIdx.x, b = bh / kH, h = bh % kH;
  const int lane = threadIdx.x & 31, nl = lane & 15, hf = lane >> 4;
  const size_t base = ((size_t)b * kH + h) * kS * kDH;
  const v8f z8 = {0.f, 0.f, 0.f, 0.f, 0.f, 0.f, 0.f, 0.f};

  v16h qa0 = load_A16x32(gq + base, kDH);
  v16h qa1 = load_A16x32(gq + base + 32, kDH);
  float mrow[8], lrow[8];
  v8f o[4];
#pragma unroll
  for (int r = 0; r < 8; ++r) { mrow[r] = NEGF; lrow[r] = 0.f; }
#pragma unroll
  for (int nt = 0; nt < 4; ++nt) o[nt] = z8;

  for (int it = 0; it < kS / 32; ++it) {
    int k0 = it * 32;
    v8f c0 = z8, c1 = z8;
    c0 = wmma_f16(qa0, load_B_tr(gk + base + (size_t)k0 * kDH, kDH), c0);
    c0 = wmma_f16(qa1, load_B_tr(gk + base + (size_t)k0 * kDH + 32, kDH), c0);
    c1 = wmma_f16(qa0, load_B_tr(gk + base + (size_t)(k0 + 16) * kDH, kDH), c1);
    c1 = wmma_f16(qa1, load_B_tr(gk + base + (size_t)(k0 + 16) * kDH + 32, kDH), c1);
    int mk0 = masked[(size_t)b * kS + k0 + nl];
    int mk1 = masked[(size_t)b * kS + k0 + 16 + nl];
#pragma unroll
    for (int r = 0; r < 8; ++r) {
      float s0 = mk0 ? NEGF : c0[r];
      float s1 = mk1 ? NEGF : c1[r];
      float mx = fmaxf(s0, s1);
#pragma unroll
      for (int off = 1; off < 16; off <<= 1) mx = fmaxf(mx, __shfl_xor(mx, off, 32));
      float mnew  = fmaxf(mrow[r], mx);
      float alpha = __expf(mrow[r] - mnew);
      float p0 = __expf(s0 - mnew);
      float p1 = __expf(s1 - mnew);
      float ps = p0 + p1;
#pragma unroll
      for (int off = 1; off < 16; off <<= 1) ps += __shfl_xor(ps, off, 32);
      lrow[r] = lrow[r] * alpha + ps;
      mrow[r] = mnew;
      Ps[(r + hf * 8) * 32 + nl]      = (_Float16)p0;
      Ps[(r + hf * 8) * 32 + nl + 16] = (_Float16)p1;
#pragma unroll
      for (int nt = 0; nt < 4; ++nt) o[nt][r] *= alpha;
    }
    asm volatile("s_wait_dscnt 0" ::: "memory");
    v16h pf = load_A16x32(Ps, 32);
#pragma unroll
    for (int nt = 0; nt < 4; ++nt)
      o[nt] = wmma_f16(pf, load_B_rm(gv + base + (size_t)k0 * kDH + nt * 16, kDH), o[nt]);
  }

#pragma unroll
  for (int nt = 0; nt < 4; ++nt)
#pragma unroll
    for (int r = 0; r < 8; ++r) {
      int g = r + hf * 8;                    // global token row (0..15)
      attn[((size_t)b * kS + g) * kE + h * kDH + nt * 16 + nl] = o[nt][r] / lrow[r];
    }
}

// ---------------------------------------------------------------- add + LN --
__global__ __launch_bounds__(256)
void add_ln_kernel(const float* __restrict__ xa, const float* __restrict__ xb,
                   const float* __restrict__ gamma, const float* __restrict__ beta,
                   float* __restrict__ outf, _Float16* __restrict__ outh) {
  __shared__ float red[16];
  int row = blockIdx.x, tid = threadIdx.x;
  float v[3];
  float s = 0.f, sq = 0.f;
#pragma unroll
  for (int j = 0; j < 3; ++j) {
    int e = tid + j * 256;
    v[j] = xa[(size_t)row * kE + e] + xb[(size_t)row * kE + e];
    s += v[j];
    sq += v[j] * v[j];
  }
#pragma unroll
  for (int off = 1; off < 32; off <<= 1) { s += __shfl_xor(s, off, 32); sq += __shfl_xor(sq, off, 32); }
  int wave = tid >> 5, lane = tid & 31;
  if (lane == 0) { red[wave] = s; red[wave + 8] = sq; }
  __syncthreads();
  float ts = 0.f, tsq = 0.f;
#pragma unroll
  for (int i = 0; i < 8; ++i) { ts += red[i]; tsq += red[i + 8]; }
  float mean = ts / kE;
  float var = tsq / kE - mean * mean;
  float inv = rsqrtf(var + 1e-5f);
#pragma unroll
  for (int j = 0; j < 3; ++j) {
    int e = tid + j * 256;
    float o = (v[j] - mean) * inv * gamma[e] + beta[e];
    if (outf) outf[(size_t)row * kE + e] = o;
    if (outh) outh[(size_t)row * kE + e] = (_Float16)o;
  }
}

// ---------------------------------------------------------------- launch ----
extern "C" void kernel_launch(void* const* d_in, const int* in_sizes, int n_in,
                              void* d_out, int out_size, void* d_ws, size_t ws_size,
                              hipStream_t stream) {
  (void)in_sizes; (void)n_in; (void)out_size; (void)ws_size;
  const float* x    = (const float*)d_in[0];
  const int*   am   = (const int*)d_in[1];
  const int*   gm   = (const int*)d_in[2];
  const float* Wq   = (const float*)d_in[3];
  const float* bq   = (const float*)d_in[4];
  const float* Wk   = (const float*)d_in[5];
  const float* bk   = (const float*)d_in[6];
  const float* Wv   = (const float*)d_in[7];
  const float* bv   = (const float*)d_in[8];
  const float* Wqg  = (const float*)d_in[9];
  const float* bqg  = (const float*)d_in[10];
  const float* Wkg  = (const float*)d_in[11];
  const float* bkg  = (const float*)d_in[12];
  const float* Wvg  = (const float*)d_in[13];
  const float* bvg  = (const float*)d_in[14];
  const float* ln0g = (const float*)d_in[15];
  const float* ln0b = (const float*)d_in[16];
  const float* ln1g = (const float*)d_in[17];
  const float* ln1b = (const float*)d_in[18];
  const float* W1   = (const float*)d_in[19];
  const float* W3   = (const float*)d_in[20];
  const float* W2   = (const float*)d_in[21];

  char* p = (char*)d_ws;
  auto alloc = [&](size_t bytes) -> void* {
    void* r = (void*)p;
    p += (bytes + 255) & ~(size_t)255;
    return r;
  };

  const size_t NTOK = (size_t)kB * kS;
  float*     t       = (float*)alloc(NTOK * 4);
  int*       removed = (int*)alloc(NTOK * 4);
  int*       masked  = (int*)alloc(NTOK * 4);
  _Float16*  xh      = (_Float16*)alloc(NTOK * kE * 2);
  _Float16*  Wh[6];
  for (int i = 0; i < 6; ++i) Wh[i] = (_Float16*)alloc((size_t)kE * kE * 2);
  _Float16*  W1h     = (_Float16*)alloc((size_t)kE * kFF * 2);
  _Float16*  W3h     = (_Float16*)alloc((size_t)kE * kFF * 2);
  _Float16*  W2h     = (_Float16*)alloc((size_t)kFF * kE * 2);
  float*     tmp     = (float*)alloc(NTOK * kE * 4);
  _Float16*  qh      = (_Float16*)alloc(NTOK * kE * 2);
  _Float16*  kh      = (_Float16*)alloc(NTOK * kE * 2);
  _Float16*  vh      = (_Float16*)alloc(NTOK * kE * 2);
  _Float16*  gqh     = (_Float16*)alloc(NTOK * kE * 2);
  _Float16*  gkh     = (_Float16*)alloc(NTOK * kE * 2);
  _Float16*  gvh     = (_Float16*)alloc(NTOK * kE * 2);
  float*     attn    = (float*)alloc(NTOK * kE * 4);
  float*     h0      = (float*)alloc(NTOK * kE * 4);
  _Float16*  h0h     = (_Float16*)alloc(NTOK * kE * 2);
  _Float16*  mid     = (_Float16*)alloc(NTOK * kFF * 2);
  float*     ffn     = (float*)alloc(NTOK * kE * 4);

  prep_kernel<<<kB, 256, 0, stream>>>(am, gm, t, removed, masked);

  const int nbig = kB * kS * kE;
  cvt_kernel<<<(nbig + 255) / 256, 256, 0, stream>>>(x, xh, nbig);
  const float* Ws[6] = {Wq, Wk, Wv, Wqg, Wkg, Wvg};
  for (int i = 0; i < 6; ++i)
    cvt_kernel<<<(kE * kE + 255) / 256, 256, 0, stream>>>(Ws[i], Wh[i], kE * kE);
  cvt_kernel<<<(kE * kFF + 255) / 256, 256, 0, stream>>>(W1, W1h, kE * kFF);
  cvt_kernel<<<(kE * kFF + 255) / 256, 256, 0, stream>>>(W3, W3h, kE * kFF);
  cvt_kernel<<<(kFF * kE + 255) / 256, 256, 0, stream>>>(W2, W2h, kFF * kE);

  const float* biases[6] = {bq, bk, bv, bqg, bkg, bvg};
  _Float16*    dsts[6]   = {qh, kh, vh, gqh, gkh, gvh};
  const int    ropes[6]  = {1, 1, 0, 0, 0, 0};
  const float  scales[6] = {0.125f, 1.f, 1.f, 0.125f, 1.f, 1.f};

  dim3 gProj((kB * kS) / 128, kE / 128);
  for (int i = 0; i < 6; ++i) {
    gemm_wmma<0><<<gProj, 256, 0, stream>>>(xh, Wh[i], nullptr, biases[i], tmp, nullptr,
                                            kB * kS, kE, kE);
    pack_kernel<<<(nbig + 255) / 256, 256, 0, stream>>>(tmp, t, dsts[i], ropes[i], scales[i]);
  }

  band_attn_kernel<<<dim3(kC, kH, kB), 256, 0, stream>>>(qh, kh, vh, removed, masked, attn);
  global_attn_kernel<<<kB * kH, 32, 0, stream>>>(gqh, gkh, gvh, masked, attn);

  add_ln_kernel<<<kB * kS, 256, 0, stream>>>(x, attn, ln0g, ln0b, h0, h0h);

  dim3 gFF1((kB * kS) / 128, kFF / 128);
  gemm_wmma<2><<<gFF1, 256, 0, stream>>>(h0h, W1h, W3h, nullptr, nullptr, mid,
                                         kB * kS, kFF, kE);
  dim3 gFF2((kB * kS) / 128, kE / 128);
  gemm_wmma<0><<<gFF2, 256, 0, stream>>>(mid, W2h, nullptr, nullptr, ffn, nullptr,
                                         kB * kS, kE, kFF);

  add_ln_kernel<<<kB * kS, 256, 0, stream>>>(h0, ffn, ln1g, ln1b, (float*)d_out, nullptr);
}
